// Multihead_75222057222155
// MI455X (gfx1250) — compile-verified, matrix operand order checked by reading, not run on hardware
//
#include <hip/hip_runtime.h>
#include <hip/hip_bf16.h>

typedef __attribute__((ext_vector_type(16))) _Float16 v16h;
typedef __attribute__((ext_vector_type(8)))  float    v8f;

#define NUM_HEAD 8
#define DQK      256
#define DV       256
#define OUTD     256
#define BB       16
#define SLEN     512
#define TOK      (BB * SLEN)          // 8192
#define CC       (NUM_HEAD * DV)      // 2048

// ---------------------------------------------------------------------------
// WMMA helpers (gfx1250, wave32).  A: 16x32 f16, one row per lane;
// element j of v16h holds K = 16*(j/8) + 8*(lane/16) + (j&7).
// B loaded identically from a row-major [N][K] source (i.e. B^T).
// C/D: v8f, element v -> (M = v + 8*(lane/16), N = lane&15).
// ---------------------------------------------------------------------------
__device__ __forceinline__ v8f wmma16(v16h a, v16h b, v8f c) {
  return __builtin_amdgcn_wmma_f32_16x16x32_f16(false, a, false, b, (short)0, c,
                                                false, false);
}

__device__ __forceinline__ v16h frag_f16(const _Float16* __restrict__ p, int ld) {
  const int lane = threadIdx.x & 31;
  const int half = lane >> 4;
  const int r    = lane & 15;
  v16h f;
#pragma unroll
  for (int j = 0; j < 16; ++j) {
    int k = ((j >> 3) << 4) + (half << 3) + (j & 7);
    f[j] = p[(size_t)r * ld + k];
  }
  return f;
}

__device__ __forceinline__ v16h frag_f32(const float* __restrict__ p, int ld) {
  const int lane = threadIdx.x & 31;
  const int half = lane >> 4;
  const int r    = lane & 15;
  v16h f;
#pragma unroll
  for (int j = 0; j < 16; ++j) {
    int k = ((j >> 3) << 4) + (half << 3) + (j & 7);
    f[j] = (_Float16)p[(size_t)r * ld + k];
  }
  return f;
}

// ---------------------------------------------------------------------------
// Stage 0: transpose + f32->f16 convert:  dst[b][c][r] = src[b][r][c]
// ---------------------------------------------------------------------------
__global__ void transpose_f16_kernel(const float* __restrict__ src,
                                     _Float16* __restrict__ dst,
                                     int batch, int rows, int cols) {
  size_t total = (size_t)batch * rows * cols;
  for (size_t i = (size_t)blockIdx.x * blockDim.x + threadIdx.x; i < total;
       i += (size_t)gridDim.x * blockDim.x) {
    int    c   = (int)(i % cols);
    size_t t   = i / cols;
    int    r   = (int)(t % rows);
    int    bch = (int)(t / rows);
    dst[((size_t)bch * cols + c) * rows + r] = (_Float16)src[i];
  }
}

// ---------------------------------------------------------------------------
// Stage 1: per-head projection  Out = X * WT^T + bias   (f16 output)
// X: [TOK][256] f32,  WT: [h][e][d] f16,  bias: [h][256] f32
// mode 0: Out[h][t][e] row-major;  mode 1: Out[h][b][e][k] (transposed store)
// One wave per 16x16 tile; grid = NUM_HEAD * 512 * 16 waves.
// ---------------------------------------------------------------------------
__global__ void proj_kernel(const float* __restrict__ X,
                            const _Float16* __restrict__ WT,
                            const float* __restrict__ bias,
                            _Float16* __restrict__ Out, int mode) {
  const int wave  = blockIdx.x * (blockDim.x >> 5) + (threadIdx.x >> 5);
  const int ntile = wave & 15;
  const int mtile = (wave >> 4) & 511;
  const int h     = wave >> 13;
  const int lane  = threadIdx.x & 31;
  const int half  = lane >> 4;
  const int col   = ntile * 16 + (lane & 15);

  const float*    Xb = X + (size_t)(mtile * 16) * DQK;
  const _Float16* Wb = WT + (size_t)h * DQK * DQK + (size_t)(ntile * 16) * DQK;

  v8f acc = {};
#pragma unroll
  for (int kk = 0; kk < 8; ++kk) {
    v16h a = frag_f32(Xb + kk * 32, DQK);
    v16h b = frag_f16(Wb + kk * 32, DQK);
    acc = wmma16(a, b, acc);
  }
  const float bv = bias[h * 256 + col];

  if (mode == 0) {
#pragma unroll
    for (int v = 0; v < 8; ++v) {
      int m = mtile * 16 + v + half * 8;
      Out[((size_t)h * TOK + m) * DQK + col] = (_Float16)(acc[v] + bv);
    }
  } else {  // transposed store for V:  VpT[h][b][dv][k]
#pragma unroll
    for (int v = 0; v < 8; ++v) {
      int t  = mtile * 16 + v + half * 8;
      int bb = t >> 9, kt = t & 511;
      Out[(((size_t)h * BB + bb) * DV + col) * SLEN + kt] = (_Float16)(acc[v] + bv);
    }
  }
}

// ---------------------------------------------------------------------------
// Stage 2: flash attention.  One wave per (h, b, 16-row q-tile).
// Qp/Kp: [h][t][e] f16,  VpT: [h][b][dv][k] f16,  Hc out: [t][h*256+dv] f16
// ---------------------------------------------------------------------------
__global__ void attn_kernel(const _Float16* __restrict__ Qp,
                            const _Float16* __restrict__ Kp,
                            const _Float16* __restrict__ VpT,
                            const int* __restrict__ Kmask,
                            _Float16* __restrict__ Hc) {
  __shared__ _Float16 plds[8 * 16 * 32];  // 1KB per wave
  const int wid   = threadIdx.x >> 5;
  const int wave  = blockIdx.x * 8 + wid;
  const int qtile = wave & 31;
  const int bb    = (wave >> 5) & 15;
  const int h     = wave >> 9;
  const int lane  = threadIdx.x & 31;
  const int half  = lane >> 4;
  const int nn    = lane & 15;
  _Float16* pl = plds + wid * 512;

  const _Float16* Qbase = Qp + ((size_t)h * TOK + bb * SLEN + qtile * 16) * DQK;
  const _Float16* Kbase = Kp + ((size_t)h * TOK + bb * SLEN) * DQK;
  const _Float16* Vbase = VpT + ((size_t)h * BB + bb) * (size_t)DV * SLEN;
  const int*      mb    = Kmask + bb * SLEN;

  float mrow[8], lrow[8];
  v8f o[16] = {};
#pragma unroll
  for (int v = 0; v < 8; ++v) { mrow[v] = -1e30f; lrow[v] = 0.0f; }

  for (int kc = 0; kc < 16; ++kc) {  // 32 key columns per iteration
    v8f s0 = {}, s1 = {};
#pragma unroll
    for (int ee = 0; ee < 8; ++ee) {
      v16h a  = frag_f16(Qbase + ee * 32, DQK);
      v16h b0 = frag_f16(Kbase + (size_t)(kc * 32) * DQK + ee * 32, DQK);
      v16h b1 = frag_f16(Kbase + (size_t)(kc * 32 + 16) * DQK + ee * 32, DQK);
      s0 = wmma16(a, b0, s0);
      s1 = wmma16(a, b1, s1);
    }
    const int mk0 = mb[kc * 32 + nn];
    const int mk1 = mb[kc * 32 + 16 + nn];
    float rmax[8];
#pragma unroll
    for (int v = 0; v < 8; ++v) {
      float x0 = s0[v] * 0.0625f; if (mk0 == 1) x0 = -1e30f;
      float x1 = s1[v] * 0.0625f; if (mk1 == 1) x1 = -1e30f;
      s0[v] = x0; s1[v] = x1;
      float r = fmaxf(x0, x1);
      r = fmaxf(r, __shfl_xor(r, 1, 32));
      r = fmaxf(r, __shfl_xor(r, 2, 32));
      r = fmaxf(r, __shfl_xor(r, 4, 32));
      r = fmaxf(r, __shfl_xor(r, 8, 32));
      rmax[v] = r;
    }
    float fs[8];
#pragma unroll
    for (int v = 0; v < 8; ++v) {
      float mn = fmaxf(mrow[v], rmax[v]);
      float f  = __expf(mrow[v] - mn);
      mrow[v]  = mn;
      float p0 = __expf(s0[v] - mn);
      float p1 = __expf(s1[v] - mn);
      s0[v] = p0; s1[v] = p1;
      float rs = p0 + p1;
      rs += __shfl_xor(rs, 1, 32);
      rs += __shfl_xor(rs, 2, 32);
      rs += __shfl_xor(rs, 4, 32);
      rs += __shfl_xor(rs, 8, 32);
      lrow[v] = lrow[v] * f + rs;
      fs[v] = f;
    }
#pragma unroll
    for (int t = 0; t < 16; ++t)
#pragma unroll
      for (int v = 0; v < 8; ++v) o[t][v] *= fs[v];

    // C-layout -> A-layout for P via this wave's LDS slab (DS in-order/wave)
#pragma unroll
    for (int v = 0; v < 8; ++v) {
      int m = v + half * 8;
      pl[m * 32 + nn]      = (_Float16)s0[v];
      pl[m * 32 + 16 + nn] = (_Float16)s1[v];
    }
    asm volatile("s_wait_dscnt 0" ::: "memory");
    v16h pa = frag_f16(pl, 32);
#pragma unroll
    for (int t = 0; t < 16; ++t) {
      v16h bv = frag_f16(Vbase + (size_t)(t * 16) * SLEN + kc * 32, SLEN);
      o[t] = wmma16(pa, bv, o[t]);
    }
    asm volatile("" ::: "memory");
  }

#pragma unroll
  for (int v = 0; v < 8; ++v) {
    float inv = 1.0f / lrow[v];
    int   t   = bb * SLEN + qtile * 16 + v + half * 8;
#pragma unroll
    for (int nt = 0; nt < 16; ++nt) {
      int c = h * DV + nt * 16 + nn;
      Hc[(size_t)t * CC + c] = (_Float16)(o[nt][v] * inv);
    }
  }
}

// ---------------------------------------------------------------------------
// Stage 3: out = Hc[TOK][2048] * Wout + bout   (f32 output)
// WoT: [o][c] f16.  One wave per 16x16 tile, K = 2048.
// ---------------------------------------------------------------------------
__global__ void out_kernel(const _Float16* __restrict__ Hc,
                           const _Float16* __restrict__ WoT,
                           const float* __restrict__ bout,
                           float* __restrict__ out) {
  const int wave  = blockIdx.x * (blockDim.x >> 5) + (threadIdx.x >> 5);
  const int ntile = wave & 15;
  const int mtile = wave >> 4;
  const int lane  = threadIdx.x & 31;
  const int half  = lane >> 4;
  const int col   = ntile * 16 + (lane & 15);

  const _Float16* Ab = Hc + (size_t)(mtile * 16) * CC;
  const _Float16* Bb = WoT + (size_t)(ntile * 16) * CC;

  v8f acc = {};
#pragma unroll 4
  for (int kk = 0; kk < 64; ++kk) {
    v16h a = frag_f16(Ab + kk * 32, CC);
    v16h b = frag_f16(Bb + kk * 32, CC);
    acc = wmma16(a, b, acc);
  }
  const float bv = bout[col];
#pragma unroll
  for (int v = 0; v < 8; ++v) {
    int t = mtile * 16 + v + half * 8;
    out[(size_t)t * OUTD + col] = acc[v] + bv;
  }
}

// ---------------------------------------------------------------------------
extern "C" void kernel_launch(void* const* d_in, const int* in_sizes, int n_in,
                              void* d_out, int out_size, void* d_ws, size_t ws_size,
                              hipStream_t stream) {
  const float* Q     = (const float*)d_in[0];
  const float* K     = (const float*)d_in[1];
  const float* V     = (const float*)d_in[2];
  const int*   Kmask = (const int*)d_in[3];
  const float* Wq    = (const float*)d_in[4];
  const float* bq    = (const float*)d_in[5];
  const float* Wk    = (const float*)d_in[6];
  const float* bk    = (const float*)d_in[7];
  const float* Wv    = (const float*)d_in[8];
  const float* bv    = (const float*)d_in[9];
  const float* Wout  = (const float*)d_in[10];
  const float* bout  = (const float*)d_in[11];
  float* out = (float*)d_out;

  char*  ws  = (char*)d_ws;
  size_t off = 0;
  auto carve = [&](size_t bytes) -> void* {
    void* p = ws + off;
    off += (bytes + 255) & ~(size_t)255;
    return p;
  };
  _Float16* WqT = (_Float16*)carve((size_t)NUM_HEAD * DQK * DQK * 2);
  _Float16* WkT = (_Float16*)carve((size_t)NUM_HEAD * DQK * DQK * 2);
  _Float16* WvT = (_Float16*)carve((size_t)NUM_HEAD * DV * DV * 2);
  _Float16* WoT = (_Float16*)carve((size_t)CC * OUTD * 2);
  _Float16* Qp  = (_Float16*)carve((size_t)NUM_HEAD * TOK * DQK * 2);
  _Float16* Kp  = (_Float16*)carve((size_t)NUM_HEAD * TOK * DQK * 2);
  _Float16* VpT = (_Float16*)carve((size_t)NUM_HEAD * TOK * DV * 2);
  _Float16* Hc  = (_Float16*)carve((size_t)TOK * CC * 2);
  (void)ws_size; (void)in_sizes; (void)n_in; (void)out_size;

  // Stage 0: weight transpose + f16 convert
  transpose_f16_kernel<<<1024, 256, 0, stream>>>(Wq, WqT, NUM_HEAD, DQK, DQK);
  transpose_f16_kernel<<<1024, 256, 0, stream>>>(Wk, WkT, NUM_HEAD, DQK, DQK);
  transpose_f16_kernel<<<1024, 256, 0, stream>>>(Wv, WvT, NUM_HEAD, DV, DV);
  transpose_f16_kernel<<<1024, 256, 0, stream>>>(Wout, WoT, 1, CC, OUTD);

  // Stage 1: projections (8 waves / block; 65536 waves each)
  proj_kernel<<<8192, 256, 0, stream>>>(Q, WqT, bq, Qp, 0);
  proj_kernel<<<8192, 256, 0, stream>>>(K, WkT, bk, Kp, 0);
  proj_kernel<<<8192, 256, 0, stream>>>(V, WvT, bv, VpT, 1);

  // Stage 2: attention (4096 waves)
  attn_kernel<<<512, 256, 0, stream>>>(Qp, Kp, VpT, Kmask, Hc);

  // Stage 3: output GEMM (8192 waves)
  out_kernel<<<1024, 256, 0, stream>>>(Hc, WoT, bout, out);
}
